// Quantizer_42923903156708
// MI455X (gfx1250) — matrix operand order repaired and checked by reading, not compiled, not run on hardware
//
#include <hip/hip_runtime.h>

typedef __attribute__((ext_vector_type(16))) _Float16 v16h;
typedef __attribute__((ext_vector_type(8)))  _Float16 v8h;
typedef __attribute__((ext_vector_type(8)))  float    v8f;

#define C_DIM       64
#define K_CODES     1024
#define N_ROWS      131072            // 8388608 / 64
#define LDS_STRIDE  72                // halves per codebook entry (144B: 16B-aligned, bank-spread)
#define NUM_WG      256
#define THREADS     256
#define WAVES       8
#define RT          4                 // row tiles blocked per wave sweep (64 rows)
#define ROW_TILES   (N_ROWS / 16)     // 8192 = NUM_WG * WAVES * RT
#define SMEM_BYTES  (K_CODES * LDS_STRIDE * 2 + K_CODES * 4)   // f16 codebook + f32 (-e2/2)

__global__ void vq_zero_loss(float* p) { p[0] = 0.0f; }

__global__ __launch_bounds__(THREADS)
void vq_quantize_kernel(const float* __restrict__ x,
                        const float* __restrict__ cb,
                        float* __restrict__ out,
                        float* __restrict__ loss)
{
    extern __shared__ char smem[];
    _Float16* sCB = (_Float16*)smem;                       // [K_CODES][LDS_STRIDE] f16
    float*    sE2 = (float*)(smem + K_CODES * LDS_STRIDE * 2);  // -0.5*||e||^2

    const int tid  = threadIdx.x;
    const int lane = tid & 31;
    const int wave = tid >> 5;
    const int lo   = lane & 15;       // column-within-tile / row-within-tile
    const int hi   = lane >> 4;       // half selector per WMMA layout

    // ---- stage codebook into LDS as f16, plus -0.5*||e||^2 ----
    for (int e = tid; e < K_CODES; e += THREADS) {
        const float4* cp = (const float4*)(cb + (size_t)e * C_DIM);
        float s2 = 0.0f;
        #pragma unroll
        for (int j = 0; j < 8; ++j) {
            float4 f0 = cp[2 * j];
            float4 f1 = cp[2 * j + 1];
            s2 += f0.x * f0.x + f0.y * f0.y + f0.z * f0.z + f0.w * f0.w;
            s2 += f1.x * f1.x + f1.y * f1.y + f1.z * f1.z + f1.w * f1.w;
            v8h h;
            h[0] = (_Float16)f0.x; h[1] = (_Float16)f0.y;
            h[2] = (_Float16)f0.z; h[3] = (_Float16)f0.w;
            h[4] = (_Float16)f1.x; h[5] = (_Float16)f1.y;
            h[6] = (_Float16)f1.z; h[7] = (_Float16)f1.w;
            *(v8h*)&sCB[e * LDS_STRIDE + 8 * j] = h;       // ds_store_b128
        }
        sE2[e] = -0.5f * s2;
    }
    __syncthreads();

    // ---- one sweep: 4 row tiles (64 rows) per wave, full codebook from LDS ----
    const int n0 = (blockIdx.x * (WAVES * RT) + wave * RT) * 16;

    // A matrices (16-bit A layout: lane l, half h -> channel h + (h>=8?8:0) + (l>=16?8:0))
    v16h a[RT][2];
    #pragma unroll
    for (int rt = 0; rt < RT; ++rt) {
        const int row = n0 + rt * 16 + lo;
        const float4* xp = (const float4*)(x + (size_t)row * C_DIM + hi * 8);
        float4 f0 = xp[0],  f1 = xp[1];    // ch hi*8 + 0..7
        float4 f2 = xp[4],  f3 = xp[5];    // +16
        float4 f4 = xp[8],  f5 = xp[9];    // +32
        float4 f6 = xp[12], f7 = xp[13];   // +48
        v16h a0, a1;
        a0[0]=(_Float16)f0.x;  a0[1]=(_Float16)f0.y;  a0[2]=(_Float16)f0.z;  a0[3]=(_Float16)f0.w;
        a0[4]=(_Float16)f1.x;  a0[5]=(_Float16)f1.y;  a0[6]=(_Float16)f1.z;  a0[7]=(_Float16)f1.w;
        a0[8]=(_Float16)f2.x;  a0[9]=(_Float16)f2.y;  a0[10]=(_Float16)f2.z; a0[11]=(_Float16)f2.w;
        a0[12]=(_Float16)f3.x; a0[13]=(_Float16)f3.y; a0[14]=(_Float16)f3.z; a0[15]=(_Float16)f3.w;
        a1[0]=(_Float16)f4.x;  a1[1]=(_Float16)f4.y;  a1[2]=(_Float16)f4.z;  a1[3]=(_Float16)f4.w;
        a1[4]=(_Float16)f5.x;  a1[5]=(_Float16)f5.y;  a1[6]=(_Float16)f5.z;  a1[7]=(_Float16)f5.w;
        a1[8]=(_Float16)f6.x;  a1[9]=(_Float16)f6.y;  a1[10]=(_Float16)f6.z; a1[11]=(_Float16)f6.w;
        a1[12]=(_Float16)f7.x; a1[13]=(_Float16)f7.y; a1[14]=(_Float16)f7.z; a1[15]=(_Float16)f7.w;
        a[rt][0] = a0;
        a[rt][1] = a1;
    }

    float best[RT][8];
    int   bidx[RT][8];
    #pragma unroll
    for (int rt = 0; rt < RT; ++rt)
        #pragma unroll
        for (int v = 0; v < 8; ++v) { best[rt][v] = -3.402823466e38f; bidx[rt][v] = 0; }

    // sweep 64 codebook column tiles; B from LDS (lane l: col = l&15, K = h + (h>=8?8:0) + hi*8)
    #pragma unroll 2
    for (int ct = 0; ct < K_CODES / 16; ++ct) {
        const int col = ct * 16 + lo;
        const _Float16* bp = &sCB[col * LDS_STRIDE + hi * 8];
        v8h bl0 = *(const v8h*)(bp);
        v8h bh0 = *(const v8h*)(bp + 16);
        v8h bl1 = *(const v8h*)(bp + 32);
        v8h bh1 = *(const v8h*)(bp + 48);
        v16h b0 = __builtin_shufflevector(bl0, bh0, 0,1,2,3,4,5,6,7,8,9,10,11,12,13,14,15);
        v16h b1 = __builtin_shufflevector(bl1, bh1, 0,1,2,3,4,5,6,7,8,9,10,11,12,13,14,15);

        // C operand pre-loaded with -e2/2: D = x.e - e2/2, argmax(D) == argmin(dist)
        const float e2n = sE2[col];
        v8f cinit;
        #pragma unroll
        for (int v = 0; v < 8; ++v) cinit[v] = e2n;

        #pragma unroll
        for (int rt = 0; rt < RT; ++rt) {
            v8f acc = __builtin_amdgcn_wmma_f32_16x16x32_f16(false, a[rt][0], false, b0,
                                                             (short)0, cinit, false, false);
            acc     = __builtin_amdgcn_wmma_f32_16x16x32_f16(false, a[rt][1], false, b1,
                                                             (short)0, acc, false, false);
            #pragma unroll
            for (int v = 0; v < 8; ++v) {
                if (acc[v] > best[rt][v]) { best[rt][v] = acc[v]; bidx[rt][v] = col; }
            }
        }
    }

    // max/argmax reduction across the 16 lanes holding each row (D layout: lane=N, vgpr=M)
    #pragma unroll
    for (int rt = 0; rt < RT; ++rt) {
        #pragma unroll
        for (int v = 0; v < 8; ++v) {
            #pragma unroll
            for (int off = 1; off < 16; off <<= 1) {
                float os = __shfl_xor(best[rt][v], off, 32);
                int   oi = __shfl_xor(bidx[rt][v], off, 32);
                if (os > best[rt][v] || (os == best[rt][v] && oi < bidx[rt][v])) {
                    best[rt][v] = os; bidx[rt][v] = oi;
                }
            }
        }
    }

    // gather winning fp32 codebook rows -> output; fuse loss = 1.25*mean((q-x)^2)
    // Two rows per iteration: lanes 0-15 handle row 2i (float4 of 4 channels each),
    // lanes 16-31 handle row 2i+1. 3 coalesced 16B vmem ops per 2 rows.
    float lsum = 0.0f;
    #pragma unroll
    for (int rt = 0; rt < RT; ++rt) {
        #pragma unroll
        for (int i = 0; i < 8; ++i) {
            // rows 2i and 2i+1 live in lane 0 (r<8) or lane 16 (r>=8) after reduction
            const int src = (i < 4) ? 0 : 16;
            const int q0 = __shfl(bidx[rt][(2 * i) & 7],     src, 32);
            const int q1 = __shfl(bidx[rt][(2 * i + 1) & 7], src, 32);
            const int q    = hi ? q1 : q0;
            const int rowr = n0 + rt * 16 + 2 * i + hi;
            const int c    = lo * 4;
            const float4* cbp = (const float4*)(cb + (size_t)q * C_DIM + c);
            const float4* xpp = (const float4*)(x + (size_t)rowr * C_DIM + c);
            float4*       op  = (float4*)(out + (size_t)rowr * C_DIM + c);
            float4 cv = cbp[0];
            float4 xv = xpp[0];
            op[0] = cv;
            float d0 = cv.x - xv.x;
            float d1 = cv.y - xv.y;
            float d2 = cv.z - xv.z;
            float d3 = cv.w - xv.w;
            lsum = __builtin_fmaf(d0, d0, lsum);
            lsum = __builtin_fmaf(d1, d1, lsum);
            lsum = __builtin_fmaf(d2, d2, lsum);
            lsum = __builtin_fmaf(d3, d3, lsum);
        }
    }
    #pragma unroll
    for (int off = 1; off < 32; off <<= 1)
        lsum += __shfl_xor(lsum, off, 32);
    if (lane == 0)
        atomicAdd(loss, lsum * (1.25f / 8388608.0f));
}

extern "C" void kernel_launch(void* const* d_in, const int* in_sizes, int n_in,
                              void* d_out, int out_size, void* d_ws, size_t ws_size,
                              hipStream_t stream) {
    const float* x  = (const float*)d_in[0];
    const float* cb = (const float*)d_in[1];
    float* out  = (float*)d_out;
    float* loss = out + (out_size - 1);   // scalar embedding_loss appended after quantized

    vq_zero_loss<<<1, 1, 0, stream>>>(loss);
    vq_quantize_kernel<<<NUM_WG, THREADS, SMEM_BYTES, stream>>>(x, cb, out, loss);
}
